// shape_align_68564857913609
// MI455X (gfx1250) — matrix operand order repaired and checked by reading, not compiled, not run on hardware
//
#include <hip/hip_runtime.h>
#include <hip/hip_bf16.h>

// ---------------- problem constants (match reference) ----------------
#define BB   4
#define CHN  256
#define HH   48
#define WW   160
#define AA   36
#define K2   9
#define KTOT (CHN * K2)      // 2304
#define HW   (HH * WW)       // 7680
#define NPIX (BB * HW)       // 30720
#define NT   32              // pixel tile (N) per workgroup
#define KC   32              // K chunk per WMMA step
#define NSTEP (KTOT / KC)    // 72

typedef __attribute__((ext_vector_type(16))) _Float16 v16h;
typedef __attribute__((ext_vector_type(8)))  float    v8f;

// 32 B: two row-pair origins + 4 fused (bilinear * validity * mask) weights
struct TapRec {
    int   oT, oB;            // top/bottom pair element offsets (cy*W + ox)
    float wT0, wT1;          // weights for x[oT], x[oT+1]
    float wB0, wB1;          // weights for x[oB], x[oB+1]
    int   pad0, pad1;
};

struct __attribute__((aligned(4))) f2 { float x, y; };
static __device__ __forceinline__ f2 ld2(const float* p) {
    f2 r; __builtin_memcpy(&r, p, 8); return r;   // b64 (or 2xb32) global load
}

static __device__ __forceinline__ int iclamp(int v, int lo, int hi) {
    return v < lo ? lo : (v > hi ? hi : v);
}

// ---------------- kernel 0: weight f32 -> f16, K permuted to tap-major ------
// src K index = c*9 + t  (weight.reshape(O, C*K2));  dst K index = t*256 + c
__global__ void wconv_kernel(const float* __restrict__ w, _Float16* __restrict__ wf) {
    int i = blockIdx.x * blockDim.x + threadIdx.x;
    if (i < CHN * KTOT) {
        int m = i / KTOT;
        int k = i - m * KTOT;
        int c = k / K2;
        int t = k - c * K2;
        wf[(size_t)m * KTOT + t * CHN + c] = (_Float16)w[i];
    }
}

// ---------------- kernel 1: top-k gate + paired tap records ----------------
__global__ void gate_kernel(const float* __restrict__ prob,
                            const float* __restrict__ table,
                            TapRec* __restrict__ taps) {
    int p = blockIdx.x * blockDim.x + threadIdx.x;
    if (p >= NPIX) return;
    int b = p / HW;
    int rem = p - b * HW;
    int y = rem / WW;
    int x = rem - y * WW;

    // top-3 over anchors (strict > keeps lowest index on ties, like lax.top_k)
    const float* pb = prob + (size_t)b * AA * HW + rem;
    float v0 = -1e30f, v1 = -1e30f, v2 = -1e30f;
    int   i0 = 0, i1 = 0, i2 = 0;
    for (int a = 0; a < AA; ++a) {
        float v = pb[(size_t)a * HW];
        if (v > v0)      { v2 = v1; i2 = i1; v1 = v0; i1 = i0; v0 = v; i0 = a; }
        else if (v > v1) { v2 = v1; i2 = i1; v1 = v;  i1 = a; }
        else if (v > v2) { v2 = v;  i2 = a; }
    }
    float e1 = __expf(v1 - v0), e2 = __expf(v2 - v0);
    float inv = 1.0f / (1.0f + e1 + e2);
    float s0 = inv, s1 = e1 * inv, s2 = e2 * inv;
    float m    = v0;                        // mask (mrep) — always applied
    float hard = (m > 0.5f) ? 1.0f : 0.0f;  // gates offsets only

    const float* t0 = table + (size_t)(i0 * 2 * K2) * HW + rem;
    const float* t1 = table + (size_t)(i1 * 2 * K2) * HW + rem;
    const float* t2 = table + (size_t)(i2 * 2 * K2) * HW + rem;

    for (int t = 0; t < K2; ++t) {
        float oh = hard * (s0 * t0[(size_t)(2 * t) * HW] +
                           s1 * t1[(size_t)(2 * t) * HW] +
                           s2 * t2[(size_t)(2 * t) * HW]);
        float ow = hard * (s0 * t0[(size_t)(2 * t + 1) * HW] +
                           s1 * t1[(size_t)(2 * t + 1) * HW] +
                           s2 * t2[(size_t)(2 * t + 1) * HW]);
        float py = (float)y + (float)(t / 3 - 1) + oh;
        float px = (float)x + (float)(t % 3 - 1) + ow;
        float fy = floorf(py), fx = floorf(px);
        float dy = py - fy,    dx = px - fx;
        int y0 = (int)fy, x0 = (int)fx, y1 = y0 + 1, x1 = x0 + 1;
        float vy0 = (y0 >= 0 && y0 < HH) ? 1.f : 0.f;
        float vy1 = (y1 >= 0 && y1 < HH) ? 1.f : 0.f;
        float vx0 = (x0 >= 0 && x0 < WW) ? 1.f : 0.f;
        float vx1 = (x1 >= 0 && x1 < WW) ? 1.f : 0.f;
        int cy0 = iclamp(y0, 0, HH - 1), cy1 = iclamp(y1, 0, HH - 1);
        int cx0 = iclamp(x0, 0, WW - 1), cx1 = iclamp(x1, 0, WW - 1);
        // corner weights, validity & mask fused
        float w00 = (1.f - dy) * (1.f - dx) * vy0 * vx0 * m;
        float w01 = (1.f - dy) * dx        * vy0 * vx1 * m;
        float w10 = dy        * (1.f - dx) * vy1 * vx0 * m;
        float w11 = dy        * dx         * vy1 * vx1 * m;
        // horizontal pairing: pair covers (ox, ox+1); scatter weights onto it.
        // Any corner whose clamped x != its pair slot has zero weight by
        // construction (validity), so the scatter is exact at all borders.
        int ox = iclamp(x0, 0, WW - 2);
        bool h0 = (cx0 - ox) != 0;          // slot of cx0 within pair
        bool h1 = (cx1 - ox) != 0;          // slot of cx1 within pair
        TapRec r;
        r.oT  = cy0 * WW + ox;
        r.oB  = cy1 * WW + ox;
        r.wT0 = (h0 ? 0.f : w00) + (h1 ? 0.f : w01);
        r.wT1 = (h0 ? w00 : 0.f) + (h1 ? w01 : 0.f);
        r.wB0 = (h0 ? 0.f : w10) + (h1 ? 0.f : w11);
        r.wB1 = (h0 ? w10 : 0.f) + (h1 ? w11 : 0.f);
        r.pad0 = 0; r.pad1 = 0;
        taps[(size_t)p * K2 + t] = r;
    }
}

// ---------------- kernel 2: fused im2col + WMMA GEMM + bias + residual ----
// A frag: 16x32 contiguous-K layout. lane<16: m=lane,    halfs {K0..15}
//                                    lane>=16: m=lane-16, halfs {K16..31}
static __device__ __forceinline__ v16h load_a16(const _Float16* row, int klo) {
    union { float4 f[2]; v16h v; } u;
    u.f[0] = *(const float4*)(row + klo);        // 8 halfs
    u.f[1] = *(const float4*)(row + klo + 8);    // 8 halfs
    return u.v;
}
// B frag: 32x16, lane<16: n=lane, K0..15 ; lane>=16: n=lane-16, K16..31
static __device__ __forceinline__ v16h load_b16(const _Float16* p) {
    union { float4 f[2]; v16h v; } u;
    u.f[0] = *(const float4*)(p);
    u.f[1] = *(const float4*)(p + 8);
    return u.v;
}
static __device__ __forceinline__ v8f wmma_f16(v16h a, v16h b, v8f c) {
    return __builtin_amdgcn_wmma_f32_16x16x32_f16(false, a, false, b,
                                                  (short)0, c, false, false);
}

__global__ __launch_bounds__(256) void dconv_gemm_kernel(
        const float*    __restrict__ xin,
        const _Float16* __restrict__ wf,
        const TapRec*   __restrict__ taps,
        const float*    __restrict__ bias,
        float*          __restrict__ out) {
    __shared__ TapRec   sTaps[NT * K2];        // 9216 B
    __shared__ _Float16 sB[2][NT * KC];        // 2 x 2048 B (double buffer)

    const int tid  = threadIdx.x;
    const int wave = tid >> 5;
    const int lane = tid & 31;

    const int pixbase = blockIdx.x * NT;      // 32 contiguous pixels, same (b,y)
    const int b   = pixbase / HW;
    const int rem = pixbase - b * HW;
    const int y   = rem / WW;
    const int xb0 = rem - y * WW;
    const float* __restrict__ xb = xin + (size_t)b * CHN * HW;  // uniform base

    for (int i = tid; i < NT * K2; i += 256)
        sTaps[i] = taps[(size_t)pixbase * K2 + i];

    v8f acc00 = {}, acc01 = {}, acc10 = {}, acc11 = {};

    const int mbase = wave * 32;
    const int mrow  = lane & 15;
    const int kl_lo = (lane < 16) ? 0 : 16;   // A half-chunk start
    const int bk_lo = (lane < 16) ? 0 : 16;   // B half start
    const int bn    = lane & 15;

    const _Float16* arow0 = wf + (size_t)(mbase + mrow) * KTOT;
    const _Float16* arow1 = arow0 + (size_t)16 * KTOT;

    const int ck  = tid & 31;                 // channel-within-chunk (= K lane)
    const int cn0 = tid >> 5;                 // pixel base (+8 per j)

    TapRec r0, r1, r2, r3;                    // tap records, fixed for 8 steps

    __syncthreads();

    // refresh tap records for tap index t (uniform per K-chunk)
    auto refresh = [&](int t) {
        r0 = sTaps[(cn0 +  0) * K2 + t];
        r1 = sTaps[(cn0 +  8) * K2 + t];
        r2 = sTaps[(cn0 + 16) * K2 + t];
        r3 = sTaps[(cn0 + 24) * K2 + t];
    };
    // build K-chunk s into dst: c = (s&7)*32 + ck, tap = s>>3 (already in r0..r3)
    auto build = [&](int s, _Float16* dst) {
        const int cOff = (((s & 7) << 5) + ck) * HW;   // 32-bit element offset
        f2 t0v = ld2(xb + cOff + r0.oT), b0v = ld2(xb + cOff + r0.oB);
        f2 t1v = ld2(xb + cOff + r1.oT), b1v = ld2(xb + cOff + r1.oB);
        f2 t2v = ld2(xb + cOff + r2.oT), b2v = ld2(xb + cOff + r2.oB);
        f2 t3v = ld2(xb + cOff + r3.oT), b3v = ld2(xb + cOff + r3.oB);
        float v0 = r0.wT0 * t0v.x + r0.wT1 * t0v.y + r0.wB0 * b0v.x + r0.wB1 * b0v.y;
        float v1 = r1.wT0 * t1v.x + r1.wT1 * t1v.y + r1.wB0 * b1v.x + r1.wB1 * b1v.y;
        float v2 = r2.wT0 * t2v.x + r2.wT1 * t2v.y + r2.wB0 * b2v.x + r2.wB1 * b2v.y;
        float v3 = r3.wT0 * t3v.x + r3.wT1 * t3v.y + r3.wB0 * b3v.x + r3.wB1 * b3v.y;
        dst[(cn0 +  0) * KC + ck] = (_Float16)v0;
        dst[(cn0 +  8) * KC + ck] = (_Float16)v1;
        dst[(cn0 + 16) * KC + ck] = (_Float16)v2;
        dst[(cn0 + 24) * KC + ck] = (_Float16)v3;
    };

    refresh(0);
    build(0, sB[0]);
    __syncthreads();

    for (int s = 0; s < NSTEP; ++s) {
        // build next chunk into the other buffer (overlaps with WMMAs below)
        if (s + 1 < NSTEP) {
            if (((s + 1) & 7) == 0) refresh((s + 1) >> 3);
            build(s + 1, sB[(s + 1) & 1]);
        }

        const int k0 = s * KC;
        if (k0 + KC < KTOT)
            __builtin_prefetch(arow0 + k0 + KC, 0, 1);   // global_prefetch_b8

        const _Float16* cur = sB[s & 1];
        v16h a0 = load_a16(arow0 + k0, kl_lo);
        v16h a1 = load_a16(arow1 + k0, kl_lo);
        v16h b0 = load_b16(cur + bn * KC + bk_lo);
        v16h b1 = load_b16(cur + (16 + bn) * KC + bk_lo);

        acc00 = wmma_f16(a0, b0, acc00);
        acc01 = wmma_f16(a0, b1, acc01);
        acc10 = wmma_f16(a1, b0, acc10);
        acc11 = wmma_f16(a1, b1, acc11);

        __syncthreads();   // one barrier per K-step (double buffered)
    }

    // ---- epilogue: D layout (16x16 f32): vgpr r, lane<16 -> M=r, lane>=16 -> M=8+r
    const int hi = lane >> 4;
    const int nn = lane & 15;
    #pragma unroll
    for (int r = 0; r < 8; ++r) {
        int m0 = mbase + r + 8 * hi;
        int m1 = m0 + 16;
        {   // acc00: (m0, nn)
            size_t idx = (((size_t)b * CHN + m0) * HH + y) * WW + xb0 + nn;
            out[idx] = acc00[r] + bias[m0] + xin[idx];
        }
        {   // acc01: (m0, 16+nn)
            size_t idx = (((size_t)b * CHN + m0) * HH + y) * WW + xb0 + 16 + nn;
            out[idx] = acc01[r] + bias[m0] + xin[idx];
        }
        {   // acc10: (m1, nn)
            size_t idx = (((size_t)b * CHN + m1) * HH + y) * WW + xb0 + nn;
            out[idx] = acc10[r] + bias[m1] + xin[idx];
        }
        {   // acc11: (m1, 16+nn)
            size_t idx = (((size_t)b * CHN + m1) * HH + y) * WW + xb0 + 16 + nn;
            out[idx] = acc11[r] + bias[m1] + xin[idx];
        }
    }
}

// ---------------- launch ----------------
extern "C" void kernel_launch(void* const* d_in, const int* in_sizes, int n_in,
                              void* d_out, int out_size, void* d_ws, size_t ws_size,
                              hipStream_t stream) {
    (void)in_sizes; (void)n_in; (void)out_size; (void)ws_size;
    const float* x      = (const float*)d_in[0];
    const float* prob   = (const float*)d_in[1];
    const float* table  = (const float*)d_in[2];
    const float* weight = (const float*)d_in[3];
    const float* bias   = (const float*)d_in[4];
    // d_in[5] = k (==3, compile-time TOPK)
    float* out = (float*)d_out;

    unsigned char* ws = (unsigned char*)d_ws;
    _Float16* wf  = (_Float16*)ws;                               // 1,179,648 B
    TapRec*   tps = (TapRec*)(ws + (size_t)CHN * KTOT * 2);      // 8,847,360 B

    wconv_kernel<<<(CHN * KTOT + 255) / 256, 256, 0, stream>>>(weight, wf);
    gate_kernel<<<(NPIX + 255) / 256, 256, 0, stream>>>(prob, table, tps);
    dconv_gemm_kernel<<<NPIX / NT, 256, 0, stream>>>(x, wf, tps, bias, out);
}
